// Standard8bit_83476984365431
// MI455X (gfx1250) — 8-bit matrix operand order repaired and checked by reading, compiled, not run on hardware
//
#include <hip/hip_runtime.h>
#include <stdint.h>

typedef int   v8i __attribute__((ext_vector_type(8)));
typedef int   v4i __attribute__((ext_vector_type(4)));
typedef int   v2i __attribute__((ext_vector_type(2)));

#define M_TOT 8192
#define K_TOT 4096
#define N_TOT 4096
#define ROT   256

__global__ __launch_bounds__(256) void hadamard_quant_kernel(
    const float* __restrict__ x,
    int8_t* __restrict__ q,
    float* __restrict__ scales)
{
    __shared__ __align__(16) uint8_t qbytes[K_TOT];
    __shared__ float wmax[8];

    const int row  = blockIdx.x;
    const int tid  = threadIdx.x;
    const int wave = tid >> 5;
    const int lane = tid & 31;
    const float* xrow = x + (size_t)row * K_TOT;

    float v[2][8];
    #pragma unroll
    for (int t = 0; t < 2; ++t) {
        const float* xb = xrow + (wave * 2 + t) * ROT;
        #pragma unroll
        for (int j = 0; j < 8; ++j) v[t][j] = xb[j * 32 + lane];

        #pragma unroll
        for (int b = 0; b < 5; ++b) {
            const int mask = 1 << b;
            #pragma unroll
            for (int j = 0; j < 8; ++j) {
                float o = __shfl_xor(v[t][j], mask, 32);
                v[t][j] = (lane & mask) ? (o - v[t][j]) : (v[t][j] + o);
            }
        }
        #pragma unroll
        for (int b = 1; b < 8; b <<= 1) {
            #pragma unroll
            for (int j = 0; j < 8; ++j) {
                if (!(j & b)) {
                    float a0 = v[t][j], b0 = v[t][j + b];
                    v[t][j]     = a0 + b0;
                    v[t][j + b] = a0 - b0;
                }
            }
        }
        #pragma unroll
        for (int j = 0; j < 8; ++j) v[t][j] *= 0.0625f;
    }

    float m = 0.0f;
    #pragma unroll
    for (int t = 0; t < 2; ++t)
        #pragma unroll
        for (int j = 0; j < 8; ++j) m = fmaxf(m, fabsf(v[t][j]));
    #pragma unroll
    for (int off = 16; off > 0; off >>= 1)
        m = fmaxf(m, __shfl_xor(m, off, 32));
    if (lane == 0) wmax[wave] = m;
    __syncthreads();
    float rmax = wmax[0];
    #pragma unroll
    for (int i = 1; i < 8; ++i) rmax = fmaxf(rmax, wmax[i]);

    const float s   = fmaxf(rmax * (1.0f / 127.0f), 1e-5f);
    const float inv = 1.0f / s;

    #pragma unroll
    for (int t = 0; t < 2; ++t) {
        const int blk = wave * 2 + t;
        #pragma unroll
        for (int j = 0; j < 8; ++j) {
            float r = __builtin_rintf(v[t][j] * inv);
            r = fminf(fmaxf(r, -127.0f), 127.0f);
            qbytes[blk * ROT + j * 32 + lane] = (uint8_t)(int)r;
        }
    }
    __syncthreads();

    v4i pk = *(const v4i*)&qbytes[tid * 16];
    *(v4i*)(q + (size_t)row * K_TOT + (uint32_t)(tid * 16)) = pk;

    if (tid == 0) scales[row] = s;
}

__global__ __launch_bounds__(256) void gemm_i8_kernel(
    const int8_t* __restrict__ q,
    const int8_t* __restrict__ w,
    const float*  __restrict__ sx,
    const float*  __restrict__ sw,
    const float*  __restrict__ bias,
    float*        __restrict__ out)
{
    const int tid  = threadIdx.x;
    const int wave = tid >> 5;
    const int lane = tid & 31;
    const int lh   = lane >> 4;
    const int ln   = lane & 15;

    const int m0 = blockIdx.y * 128 + (wave >> 2) * 64;
    const int n0 = blockIdx.x * 256 + (wave & 3) * 64;

    uint32_t aoff[4], boff[4];
    #pragma unroll
    for (int mt = 0; mt < 4; ++mt)
        aoff[mt] = (uint32_t)(m0 + mt * 16 + ln) * K_TOT + (uint32_t)(lh * 16);
    #pragma unroll
    for (int nt = 0; nt < 4; ++nt)
        boff[nt] = (uint32_t)(n0 + nt * 16 + ln) * K_TOT + (uint32_t)(lh * 16);

    v8i acc[4][4] = {};

    for (int k0 = 0; k0 < K_TOT; k0 += 64) {
        const int8_t* qk = q + k0;
        const int8_t* wk = w + k0;

        v8i bf[4];
        #pragma unroll
        for (int nt = 0; nt < 4; ++nt) {
            v4i lo = *(const v4i*)(wk + boff[nt]);
            v4i hi = *(const v4i*)(wk + boff[nt] + 32);
            v8i t;
            t[0] = lo[0]; t[1] = lo[1]; t[2] = lo[2]; t[3] = lo[3];
            t[4] = hi[0]; t[5] = hi[1]; t[6] = hi[2]; t[7] = hi[3];
            bf[nt] = t;
        }
        #pragma unroll
        for (int mt = 0; mt < 4; ++mt) {
            v8i af;
            #pragma unroll
            for (int p = 0; p < 4; ++p) {
                v2i d = *(const v2i*)(qk + aoff[mt] + (p & 1) * 8 + (p >> 1) * 32);
                af[2 * p]     = d[0];
                af[2 * p + 1] = d[1];
            }
            #pragma unroll
            for (int nt = 0; nt < 4; ++nt) {
                acc[mt][nt] = __builtin_amdgcn_wmma_i32_16x16x64_iu8(
                     true, af,  true, bf[nt],
                    acc[mt][nt],  false,  false);
            }
        }
    }

    #pragma unroll
    for (int mt = 0; mt < 4; ++mt) {
        #pragma unroll
        for (int r = 0; r < 8; ++r) {
            const int   mrow = m0 + mt * 16 + r + lh * 8;
            const float srow = sx[(uint32_t)mrow];
            #pragma unroll
            for (int nt = 0; nt < 4; ++nt) {
                const uint32_t ncol = (uint32_t)(n0 + nt * 16 + ln);
                const uint32_t ooff = (uint32_t)mrow * N_TOT + ncol;
                out[ooff] = (float)acc[mt][nt][r] * srow * sw[ncol] + bias[ncol];
            }
        }
    }
}

extern "C" void kernel_launch(void* const* d_in, const int* in_sizes, int n_in,
                              void* d_out, int out_size, void* d_ws, size_t ws_size,
                              hipStream_t stream) {
    const float*  x    = (const float*)d_in[0];
    const int8_t* wgt  = (const int8_t*)d_in[1];
    const float*  wsc  = (const float*)d_in[2];
    const float*  bias = (const float*)d_in[3];
    float* out = (float*)d_out;

    int8_t* q  = (int8_t*)d_ws;
    float*  sx = (float*)((char*)d_ws + (size_t)M_TOT * K_TOT);

    hadamard_quant_kernel<<<M_TOT, 256, 0, stream>>>(x, q, sx);

    dim3 grid(N_TOT / 256, M_TOT / 128);
    gemm_i8_kernel<<<grid, 256, 0, stream>>>(q, wgt, sx, wsc, bias, out);
}
